// Style_SA_18717467476188
// MI455X (gfx1250) — matrix operand order repaired and checked by reading, not compiled, not run on hardware
//
#include <hip/hip_runtime.h>

// ---------------------------------------------------------------------------
// Style self-attention (channel attention) for MI455X / gfx1250, wave32.
// Heavy GEMMs use v_wmma_f32_16x16x32_bf16 (bf16 in, f32 accumulate).
// energy is computed via the Gram-matrix identity:
//   E = Wf G Wg^T + (Wf s) bg^T + bf (Wg s)^T + N bf bg^T,  G = x x^T, s = N*mean
// Instance-norm is folded into per-batch h-conv weights Wh' = wh * rstd.
// ---------------------------------------------------------------------------

typedef __attribute__((ext_vector_type(8)))  float   v8f;
typedef __attribute__((ext_vector_type(4)))  float   v4f;
typedef __attribute__((ext_vector_type(16))) __bf16  v16bf;
typedef __attribute__((ext_vector_type(8)))  __bf16  v8bf;
typedef __attribute__((ext_vector_type(4)))  __bf16  v4bf;

#define DI static __device__ __forceinline__

constexpr int B_ = 8;
constexpr int C_ = 256;
constexpr int N_ = 16384;            // 128*128
constexpr float EPS_ = 1e-5f;

// A-operand (16x32 bf16): lane l<16 -> row l, K = {0..7, 16..23}; lanes 16..31
// -> same rows, K = {8..15, 24..31}. Per lane: two contiguous 16B chunks.
DI v16bf load_a16(const __bf16* base) {
  v8bf lo = *(const v8bf*)(base);
  v8bf hi = *(const v8bf*)(base + 16);
  return __builtin_shufflevector(lo, hi, 0, 1, 2, 3, 4, 5, 6, 7,
                                 8, 9, 10, 11, 12, 13, 14, 15);
}

DI v8f wmma_bf16(v16bf a, v16bf b, v8f c) {
  // D = A(16x32) * B(32x16) + C, f32 accumulate
  return __builtin_amdgcn_wmma_f32_16x16x32_bf16(false, a, false, b,
                                                 (short)0, c, false, false);
}

// ---------------------------------------------------------------------------
// K0: per (b,c) row stats over N + bf16 copy of x (row-major [C,N])
// ---------------------------------------------------------------------------
__global__ __launch_bounds__(256) void k_stats(const float* __restrict__ x,
                                               __bf16* __restrict__ xbf,
                                               float* __restrict__ mean,
                                               float* __restrict__ rstd) {
  int row = blockIdx.x;                       // b*C + c
  const float* xr = x + (size_t)row * N_;
  __bf16* ob = xbf + (size_t)row * N_;
  int t = threadIdx.x;
  float s = 0.f, s2 = 0.f;
  for (int i = t; i < N_ / 4; i += 256) {
    v4f v = *(const v4f*)(xr + i * 4);
    s  += v.x + v.y + v.z + v.w;
    s2 += v.x * v.x + v.y * v.y + v.z * v.z + v.w * v.w;
    *(v4bf*)(ob + i * 4) = __builtin_convertvector(v, v4bf);
  }
  __shared__ float r1[256], r2[256];
  r1[t] = s; r2[t] = s2;
  __syncthreads();
  for (int off = 128; off > 0; off >>= 1) {
    if (t < off) { r1[t] += r1[t + off]; r2[t] += r2[t + off]; }
    __syncthreads();
  }
  if (t == 0) {
    float m = r1[0] / (float)N_;
    float var = r2[0] / (float)N_ - m * m;
    mean[row] = m;
    rstd[row] = rsqrtf(var + EPS_);
  }
}

// ---------------------------------------------------------------------------
// K1: Xt[b][n][c] = (bf16) x[b][c][n]   (64x64 tiles via LDS)
// ---------------------------------------------------------------------------
__global__ __launch_bounds__(256) void k_transpose(const float* __restrict__ x,
                                                   __bf16* __restrict__ xt) {
  __shared__ __bf16 tile[64][66];
  int bid = blockIdx.x;
  int nt = bid & 255, ct = (bid >> 8) & 3, b = bid >> 10;
  int n0 = nt * 64, c0 = ct * 64;
  int tx = threadIdx.x & 63, ty = threadIdx.x >> 6;   // 64 x 4
  const float* xb = x + (size_t)b * C_ * N_;
#pragma unroll
  for (int i = 0; i < 16; i++) {
    int c = ty + i * 4;
    tile[c][tx] = (__bf16)xb[(size_t)(c0 + c) * N_ + n0 + tx];
  }
  __syncthreads();
  __bf16* xtb = xt + (size_t)b * N_ * C_;
#pragma unroll
  for (int i = 0; i < 16; i++) {
    int n = ty + i * 4;
    xtb[(size_t)(n0 + n) * C_ + c0 + tx] = tile[tx][n];
  }
}

// ---------------------------------------------------------------------------
// K2: per (b,o): Wh'[b,o,:] = wh[o,:]*rstd[b,:] (bf16),
//     bh'[b,o] = bh[o] - sum_c wh[o,c]*mean*rstd,
//     fs[b,o] = sum_c wf[o,c]*N*mean[b,c],  gs[b,o] = sum_c wg[o,c]*N*mean[b,c]
// ---------------------------------------------------------------------------
__global__ __launch_bounds__(256) void k_wprep(const float* __restrict__ wh,
                                               const float* __restrict__ bh,
                                               const float* __restrict__ wf,
                                               const float* __restrict__ wg,
                                               const float* __restrict__ mean,
                                               const float* __restrict__ rstd,
                                               __bf16* __restrict__ whp,
                                               float* __restrict__ bhp,
                                               float* __restrict__ fs,
                                               float* __restrict__ gs) {
  int bid = blockIdx.x;
  int b = bid >> 8, o = bid & 255;
  int c = threadIdx.x;
  float m = mean[b * C_ + c], r = rstd[b * C_ + c];
  float w = wh[o * C_ + c];
  whp[((size_t)(b * C_ + o)) * C_ + c] = (__bf16)(w * r);
  float s = m * (float)N_;
  __shared__ float r1[256], r2[256], r3[256];
  r1[c] = w * m * r;
  r2[c] = wf[o * C_ + c] * s;
  r3[c] = wg[o * C_ + c] * s;
  __syncthreads();
  for (int off = 128; off > 0; off >>= 1) {
    if (c < off) { r1[c] += r1[c + off]; r2[c] += r2[c + off]; r3[c] += r3[c + off]; }
    __syncthreads();
  }
  if (c == 0) {
    bhp[b * C_ + o] = bh[o] - r1[0];
    fs[b * C_ + o] = r2[0];
    gs[b * C_ + o] = r3[0];
  }
}

// ---------------------------------------------------------------------------
// K3: f32 -> bf16 convert (for Wo)
// ---------------------------------------------------------------------------
__global__ __launch_bounds__(256) void k_cvt_w(const float* __restrict__ w,
                                               __bf16* __restrict__ wb) {
  int i = blockIdx.x * 256 + threadIdx.x;
  wb[i] = (__bf16)w[i];
}

// ---------------------------------------------------------------------------
// K4: Gram partials  Gpart[ks][b][c][d] = sum_{n in ks-slice} x[c,n]*x[d,n]
// wave = 2x2 tiles (32c x 32d), K split 4 ways. bf16 WMMA, f32 acc.
// ---------------------------------------------------------------------------
__global__ __launch_bounds__(256) void k_gram(const __bf16* __restrict__ xbf,
                                              float* __restrict__ gpart) {
  int gw = blockIdx.x * 8 + (threadIdx.x >> 5);
  int lane = threadIdx.x & 31, lh = lane >> 4, lx = lane & 15;
  int ks = gw & 3, dg = (gw >> 2) & 7, cg = (gw >> 5) & 7, b = gw >> 8;
  const __bf16* xb = xbf + (size_t)b * C_ * N_;
  const __bf16* ar0 = xb + (size_t)(cg * 32 + lx) * N_;
  const __bf16* ar1 = ar0 + (size_t)16 * N_;
  const __bf16* br0 = xb + (size_t)(dg * 32 + lx) * N_;
  const __bf16* br1 = br0 + (size_t)16 * N_;
  int k0 = ks * 4096;
  v8f a00 = {}, a01 = {}, a10 = {}, a11 = {};
  for (int k = k0; k < k0 + 4096; k += 32) {
    int ao = k + lh * 16;
    int bo = k + lh * 16;
    __builtin_prefetch(ar0 + ao + 256, 0, 1);
    __builtin_prefetch(br0 + bo + 256, 0, 1);
    v16bf a0 = *(const v16bf*)(ar0 + ao);
    v16bf a1 = *(const v16bf*)(ar1 + ao);
    v16bf b0 = *(const v16bf*)(br0 + bo);
    v16bf b1 = *(const v16bf*)(br1 + bo);
    a00 = wmma_bf16(a0, b0, a00);
    a01 = wmma_bf16(a0, b1, a01);
    a10 = wmma_bf16(a1, b0, a10);
    a11 = wmma_bf16(a1, b1, a11);
  }
  float* gp = gpart + (size_t)ks * (B_ * C_ * C_) + (size_t)b * C_ * C_;
  int cb = cg * 32 + lh * 8;
  int db = dg * 32 + lx;
#pragma unroll
  for (int r = 0; r < 8; r++) {
    gp[(cb + r) * C_ + db]           = a00[r];
    gp[(cb + r) * C_ + db + 16]      = a01[r];
    gp[(cb + 16 + r) * C_ + db]      = a10[r];
    gp[(cb + 16 + r) * C_ + db + 16] = a11[r];
  }
}

// K5: reduce 4 Gram partials (deterministic, no atomics)
__global__ __launch_bounds__(256) void k_gsum(const float* __restrict__ gp,
                                              float* __restrict__ g) {
  int i = blockIdx.x * 256 + threadIdx.x;
  const int S = B_ * C_ * C_;
  g[i] = (gp[i] + gp[i + S]) + (gp[i + 2 * S] + gp[i + 3 * S]);
}

// ---------------------------------------------------------------------------
// K6: Tt[b][d][p] = sum_q G[b][p][q] * wg[d][q]   (tiny f32 GEMM, 134 MFLOP)
// ---------------------------------------------------------------------------
__global__ __launch_bounds__(256) void k_tmat(const float* __restrict__ g,
                                              const float* __restrict__ wg,
                                              float* __restrict__ tt) {
  int idx = blockIdx.x * 256 + threadIdx.x;
  int pp = idx & 255, d = (idx >> 8) & 255, b = idx >> 16;
  const float* gr = g + ((size_t)b * C_ + pp) * C_;
  const float* wr = wg + d * C_;
  float s = 0.f;
#pragma unroll 4
  for (int q = 0; q < C_; q += 4) {
    v4f a = *(const v4f*)(gr + q);
    v4f w = *(const v4f*)(wr + q);
    s += a.x * w.x + a.y * w.y + a.z * w.z + a.w * w.w;
  }
  tt[((size_t)b * C_ + d) * C_ + pp] = s;
}

// K7: E[b][c][d] = dot(wf[c,:], Tt[b,d,:]) + fs[b,c]*bg[d] + bf[c]*gs[b,d]
//                  + N*bf[c]*bg[d]
__global__ __launch_bounds__(256) void k_energy(const float* __restrict__ wf,
                                                const float* __restrict__ tt,
                                                const float* __restrict__ bfv,
                                                const float* __restrict__ bgv,
                                                const float* __restrict__ fs,
                                                const float* __restrict__ gs,
                                                float* __restrict__ e) {
  int idx = blockIdx.x * 256 + threadIdx.x;
  int d = idx & 255, c = (idx >> 8) & 255, b = idx >> 16;
  const float* wr = wf + c * C_;
  const float* tr = tt + ((size_t)b * C_ + d) * C_;
  float s = 0.f;
#pragma unroll 4
  for (int q = 0; q < C_; q += 4) {
    v4f a = *(const v4f*)(wr + q);
    v4f t = *(const v4f*)(tr + q);
    s += a.x * t.x + a.y * t.y + a.z * t.z + a.w * t.w;
  }
  s += fs[b * C_ + c] * bgv[d] + bfv[c] * gs[b * C_ + d] +
       (float)N_ * bfv[c] * bgv[d];
  e[((size_t)b * C_ + c) * C_ + d] = s;
}

// K8: softmax over d for each (b,c) row; write TRANSPOSED bf16: attT[b][d][c]
__global__ __launch_bounds__(256) void k_softmax(const float* __restrict__ e,
                                                 __bf16* __restrict__ attT) {
  int row = blockIdx.x;            // b*C + c
  int b = row >> 8, c = row & 255;
  int d = threadIdx.x;
  float v = e[(size_t)row * C_ + d];
  __shared__ float red[256];
  red[d] = v;
  __syncthreads();
  for (int off = 128; off > 0; off >>= 1) {
    if (d < off) red[d] = fmaxf(red[d], red[d + off]);
    __syncthreads();
  }
  float mx = red[0];
  __syncthreads();
  float ex = __expf(v - mx);
  red[d] = ex;
  __syncthreads();
  for (int off = 128; off > 0; off >>= 1) {
    if (d < off) red[d] += red[d + off];
    __syncthreads();
  }
  attT[((size_t)b * C_ + d) * C_ + c] = (__bf16)(ex / red[0]);
}

// ---------------------------------------------------------------------------
// K9: out[b][n][o] = sum_c act[b][n][c] * wgt[b][o][c] (+ bias[b][o])
// act,out: [B][N][C] bf16 ; wgt per-batch [C][C] bf16. Wave: 2 n-tiles x 4
// o-tiles, K=256 fully unrolled -> 64 WMMAs straight-line.
// Used for Ht (wgt=Wh', bias=bh') and Ot (wgt=attT, no bias).
// ---------------------------------------------------------------------------
__global__ __launch_bounds__(256) void k_gemm_nc(const __bf16* __restrict__ act,
                                                 const __bf16* __restrict__ wgt,
                                                 const float* __restrict__ bias,
                                                 __bf16* __restrict__ out) {
  int gw = blockIdx.x * 8 + (threadIdx.x >> 5);
  int lane = threadIdx.x & 31, lh = lane >> 4, lx = lane & 15;
  int og = gw & 3, ng = (gw >> 2) & 511, b = gw >> 11;
  int n0 = ng * 32, o0 = og * 64;
  const __bf16* ab = act + (size_t)b * N_ * C_;
  const __bf16* wb = wgt + (size_t)b * C_ * C_;
  const __bf16* ar0 = ab + (size_t)(n0 + lx) * C_;
  const __bf16* ar1 = ar0 + 16 * C_;
  const __bf16* br[4];
#pragma unroll
  for (int oi = 0; oi < 4; oi++) br[oi] = wb + (size_t)(o0 + oi * 16 + lx) * C_;
  v8f acc[2][4] = {};
#pragma unroll
  for (int k = 0; k < C_; k += 32) {
    v16bf a0 = *(const v16bf*)(ar0 + k + lh * 16);
    v16bf a1 = *(const v16bf*)(ar1 + k + lh * 16);
#pragma unroll
    for (int oi = 0; oi < 4; oi++) {
      v16bf bm = *(const v16bf*)(br[oi] + k + lh * 16);
      acc[0][oi] = wmma_bf16(a0, bm, acc[0][oi]);
      acc[1][oi] = wmma_bf16(a1, bm, acc[1][oi]);
    }
  }
  __bf16* ob = out + (size_t)b * N_ * C_;
#pragma unroll
  for (int oi = 0; oi < 4; oi++) {
    int o = o0 + oi * 16 + lx;
    float bv = bias ? bias[b * C_ + o] : 0.f;
#pragma unroll
    for (int ni = 0; ni < 2; ni++) {
      int nb = n0 + ni * 16 + lh * 8;
#pragma unroll
      for (int r = 0; r < 8; r++)
        ob[(size_t)(nb + r) * C_ + o] = (__bf16)(acc[ni][oi][r] + bv);
    }
  }
}

// ---------------------------------------------------------------------------
// K10: final conv + bias + residual:
//   y[b][o][n] = sum_d wo[o][d]*act[b][n][d] + bo[o] + x[b][o][n]
// Output [C][N] f32, lane = n -> coalesced 64B stores.
// ---------------------------------------------------------------------------
__global__ __launch_bounds__(256) void k_gemm_cn(const __bf16* __restrict__ wob,
                                                 const __bf16* __restrict__ act,
                                                 const float* __restrict__ bias,
                                                 const float* __restrict__ resid,
                                                 float* __restrict__ out) {
  int gw = blockIdx.x * 8 + (threadIdx.x >> 5);
  int lane = threadIdx.x & 31, lh = lane >> 4, lx = lane & 15;
  int og = gw & 3, ng = (gw >> 2) & 511, b = gw >> 11;
  int n0 = ng * 32, o0 = og * 64;
  const __bf16* ab = act + (size_t)b * N_ * C_;
  const __bf16* ar[4];
#pragma unroll
  for (int oi = 0; oi < 4; oi++) ar[oi] = wob + (size_t)(o0 + oi * 16 + lx) * C_;
  const __bf16* br0 = ab + (size_t)(n0 + lx) * C_;
  const __bf16* br1 = br0 + 16 * C_;
  v8f acc[4][2] = {};
#pragma unroll
  for (int k = 0; k < C_; k += 32) {
    v16bf b0 = *(const v16bf*)(br0 + k + lh * 16);
    v16bf b1 = *(const v16bf*)(br1 + k + lh * 16);
#pragma unroll
    for (int oi = 0; oi < 4; oi++) {
      v16bf a = *(const v16bf*)(ar[oi] + k + lh * 16);
      acc[oi][0] = wmma_bf16(a, b0, acc[oi][0]);
      acc[oi][1] = wmma_bf16(a, b1, acc[oi][1]);
    }
  }
  const float* rb = resid + (size_t)b * C_ * N_;
  float* ob = out + (size_t)b * C_ * N_;
#pragma unroll
  for (int oi = 0; oi < 4; oi++) {
#pragma unroll
    for (int r = 0; r < 8; r++) {
      int o = o0 + oi * 16 + lh * 8 + r;
      float bv = bias[o];
#pragma unroll
      for (int ni = 0; ni < 2; ni++) {
        size_t off = (size_t)o * N_ + (n0 + ni * 16 + lx);
        ob[off] = acc[oi][ni][r] + bv + rb[off];
      }
    }
  }
}

// ---------------------------------------------------------------------------
extern "C" void kernel_launch(void* const* d_in, const int* in_sizes, int n_in,
                              void* d_out, int out_size, void* d_ws,
                              size_t ws_size, hipStream_t stream) {
  const float* x  = (const float*)d_in[0];
  const float* wf = (const float*)d_in[1];
  const float* bfp = (const float*)d_in[2];
  const float* wg = (const float*)d_in[3];
  const float* bgp = (const float*)d_in[4];
  const float* wh = (const float*)d_in[5];
  const float* bh = (const float*)d_in[6];
  const float* wo = (const float*)d_in[7];
  const float* bo = (const float*)d_in[8];
  float* out = (float*)d_out;

  const size_t BCN = (size_t)B_ * C_ * N_;   // 33,554,432
  const size_t BCC = (size_t)B_ * C_ * C_;   // 524,288

  char* base = (char*)d_ws;
  size_t off = 0;
  auto take = [&](size_t bytes) -> char* {
    char* r = base + off;
    off += bytes;
    off = (off + 255) & ~(size_t)255;
    return r;
  };
  float*  mean  = (float*)take(2048 * 4);
  float*  rstd  = (float*)take(2048 * 4);
  float*  bhp   = (float*)take(2048 * 4);
  float*  fs    = (float*)take(2048 * 4);
  float*  gs    = (float*)take(2048 * 4);
  __bf16* xbf   = (__bf16*)take(BCN * 2);
  __bf16* xt    = (__bf16*)take(BCN * 2);
  __bf16* whp   = (__bf16*)take(BCC * 2);
  __bf16* wob   = (__bf16*)take((size_t)C_ * C_ * 2);
  float*  gpart = (float*)take(4 * BCC * 4);
  float*  g     = (float*)take(BCC * 4);
  float*  tt    = (float*)take(BCC * 4);
  float*  e     = (float*)take(BCC * 4);
  __bf16* attT  = (__bf16*)take(BCC * 2);
  __bf16* ht    = (__bf16*)take(BCN * 2);
  __bf16* ot    = (__bf16*)take(BCN * 2);
  (void)ws_size;   // requires ~271 MB of scratch

  // 1) stats + bf16 copy of x
  k_stats<<<B_ * C_, 256, 0, stream>>>(x, xbf, mean, rstd);
  // 2) transpose to [N,C] bf16
  k_transpose<<<B_ * 4 * 256, 256, 0, stream>>>(x, xt);
  // 3) folded instance-norm weights + bias-term dots
  k_wprep<<<B_ * C_, 256, 0, stream>>>(wh, bh, wf, wg, mean, rstd,
                                       whp, bhp, fs, gs);
  // 4) Wo -> bf16
  k_cvt_w<<<C_ * C_ / 256, 256, 0, stream>>>(wo, wob);
  // 5) Gram partials (WMMA) + 6) reduce
  k_gram<<<256, 256, 0, stream>>>(xbf, gpart);
  k_gsum<<<(int)(BCC / 256), 256, 0, stream>>>(gpart, g);
  // 7) T = G Wg^T (stored transposed), 8) E = Wf T + bias terms
  k_tmat<<<(int)(BCC / 256), 256, 0, stream>>>(g, wg, tt);
  k_energy<<<(int)(BCC / 256), 256, 0, stream>>>(wf, tt, bfp, bgp, fs, gs, e);
  // 9) softmax rows -> attT (bf16, transposed)
  k_softmax<<<B_ * C_, 256, 0, stream>>>(e, attT);
  // 10) Ht[n,o] = Xt @ Wh'^T + bh'   (WMMA)
  k_gemm_nc<<<2048, 256, 0, stream>>>(xt, whp, bhp, ht);
  // 11) Ot[n,d] = Ht @ att           (WMMA)
  k_gemm_nc<<<2048, 256, 0, stream>>>(ht, attT, nullptr, ot);
  // 12) y[o,n] = Wo @ Ot^T + bo + x  (WMMA, coalesced f32 stores)
  k_gemm_cn<<<2048, 256, 0, stream>>>(wob, ot, bo, x, out);
}
